// VisionTransformer_AR_Generator_36086315221075
// MI455X (gfx1250) — matrix operand order repaired and checked by reading, not compiled, not run on hardware
//
#include <hip/hip_runtime.h>
#include <hip/hip_bf16.h>

// ---------------- problem constants (from reference setup_inputs) ----------
#define BB   8
#define CC   3
#define HH   384
#define WW   384
#define PP   16
#define EE   768
#define NLL  8
#define FFF  3072
#define LL   576            // (384/16)^2
#define NHH  12
#define HD   64             // EE / NHH
#define CPPV 768            // CC*PP*PP
#define GW   24             // HH/PP

// ---------------- WMMA types ----------------------------------------------
typedef __attribute__((ext_vector_type(16))) __bf16 v16bf;
typedef __attribute__((ext_vector_type(4)))  __bf16 bf16x4;
typedef __attribute__((ext_vector_type(8)))  float  v8f;
typedef __attribute__((ext_vector_type(4)))  float  f32x4;

union FragU { v16bf v; unsigned u[8]; };

#define BM 128
#define BN 128
#define BK 32
#define LDP (BK + 8)   // padded LDS pitch in bf16 (80B rows -> 8B aligned cols)

// ---------------------------------------------------------------------------
// Templated batched GEMM: C = act(alpha * A*B + bias) + residual
//   A: (M,K) f32 row-major.  B: (K,N) f32 if !TRANSB, (N,K) if TRANSB.
//   REQUIRES: K % 32 == 0, lda/ldb % 4 == 0, N % 4 == 0 (true for all calls).
//   Double-buffered LDS, bf16 WMMA f32-accum, 8 waves (wave32) per block.
//   batch base = (z/innerCount)*Out + (z%innerCount)*In
// ---------------------------------------------------------------------------
template<int TRANSB, int ACT, int HASBIAS, int HASRES>
__global__ __launch_bounds__(256)
void gemm_wmma_bf16(const float* __restrict__ A, int lda, long long aOut, long long aIn,
                    const float* __restrict__ Bm, int ldb, long long bOut, long long bIn,
                    float* __restrict__ C, int ldc, long long cOut, long long cIn,
                    const float* __restrict__ bias,
                    const float* __restrict__ resid, int ldr, long long rOut, long long rIn,
                    int M, int N, int K, float alpha, int innerCount)
{
    __shared__ __bf16 As[2][BM][LDP];
    __shared__ __bf16 Bs[2][BN][LDP];   // stored [n][k]

    const int tid   = threadIdx.x;
    const int lane  = tid & 31;
    const int wave  = tid >> 5;
    const int wm    = wave >> 1;     // 0..3  (M dir)
    const int wn    = wave & 1;      // 0..1  (N dir)
    const int laneM = lane & 15;
    const int hi    = lane >> 4;

    const int z  = blockIdx.z;
    const int zo = z / innerCount, zi = z % innerCount;
    const long long aBase = (long long)zo * aOut + (long long)zi * aIn;
    const long long bBase = (long long)zo * bOut + (long long)zi * bIn;
    const long long cBase = (long long)zo * cOut + (long long)zi * cIn;
    const long long rBase = (long long)zo * rOut + (long long)zi * rIn;

    const int m0 = blockIdx.y * BM;
    const int n0 = blockIdx.x * BN;

    v8f acc[2][4];
#pragma unroll
    for (int i = 0; i < 2; ++i)
#pragma unroll
        for (int j = 0; j < 4; ++j)
#pragma unroll
            for (int r = 0; r < 8; ++r) acc[i][j][r] = 0.0f;

    // ---- staging: float4 global loads, packed bf16 LDS stores -----------
    auto stage = [&](int buf, int k0) {
#pragma unroll
        for (int v = 0; v < 4; ++v) {
            const int e = (tid + v * 256) * 4;     // element id, multiple of 4
            // A tile: [row][col], col along K (contiguous both ends)
            {
                const int ar = e >> 5, ac = e & 31;
                f32x4 f = {0.f, 0.f, 0.f, 0.f};
                const int gm = m0 + ar;
                if (gm < M)
                    f = *(const f32x4*)&A[aBase + (long long)gm * lda + (k0 + ac)];
                bf16x4 b;
                b[0] = (__bf16)f[0]; b[1] = (__bf16)f[1];
                b[2] = (__bf16)f[2]; b[3] = (__bf16)f[3];
                *(bf16x4*)&As[buf][ar][ac] = b;
            }
            if (TRANSB) {
                // B is (N,K): K contiguous in memory AND in LDS -> fully vector
                const int bn = e >> 5, bc = e & 31;
                f32x4 f = {0.f, 0.f, 0.f, 0.f};
                const int gn = n0 + bn;
                if (gn < N)
                    f = *(const f32x4*)&Bm[bBase + (long long)gn * ldb + (k0 + bc)];
                bf16x4 b;
                b[0] = (__bf16)f[0]; b[1] = (__bf16)f[1];
                b[2] = (__bf16)f[2]; b[3] = (__bf16)f[3];
                *(bf16x4*)&Bs[buf][bn][bc] = b;
            } else {
                // B is (K,N): vector load along N, scatter-transpose into [n][k]
                const int bk = e >> 7, bn = e & 127;
                f32x4 f = {0.f, 0.f, 0.f, 0.f};
                if (n0 + bn < N)   // bn%4==0 && N%4==0 -> whole float4 in/out
                    f = *(const f32x4*)&Bm[bBase + (long long)(k0 + bk) * ldb + (n0 + bn)];
                Bs[buf][bn + 0][bk] = (__bf16)f[0];
                Bs[buf][bn + 1][bk] = (__bf16)f[1];
                Bs[buf][bn + 2][bk] = (__bf16)f[2];
                Bs[buf][bn + 3][bk] = (__bf16)f[3];
            }
        }
    };

    // ---- fragment math ---------------------------------------------------
    auto compute = [&](int buf) {
        FragU fb[4];
#pragma unroll
        for (int nf = 0; nf < 4; ++nf) {
            const unsigned* brow = (const unsigned*)&Bs[buf][wn * 64 + nf * 16 + laneM][0];
            const int kb = hi * 8;     // K 0-15 (lanes 0-15) / 16-31 (lanes 16-31)
#pragma unroll
            for (int j = 0; j < 8; ++j) fb[nf].u[j] = brow[kb + j];
        }
#pragma unroll
        for (int mf = 0; mf < 2; ++mf) {
            FragU fa;
            const unsigned* arow = (const unsigned*)&As[buf][wm * 32 + mf * 16 + laneM][0];
            const int kb = hi * 8;     // K 0-15 (lanes 0-15) / 16-31 (lanes 16-31)
#pragma unroll
            for (int j = 0; j < 8; ++j) fa.u[j] = arow[kb + j];
#pragma unroll
            for (int nf = 0; nf < 4; ++nf)
                acc[mf][nf] = __builtin_amdgcn_wmma_f32_16x16x32_bf16(
                    false, fa.v, false, fb[nf].v, (short)0, acc[mf][nf], false, false);
        }
    };

    // ---- double-buffered main loop: one barrier per K-step --------------
    const int nk = K >> 5;             // K % 32 == 0 guaranteed
    stage(0, 0);
    __syncthreads();
    for (int kt = 0; kt < nk; ++kt) {
        const int cur = kt & 1;
        if (kt + 1 < nk) {
            if (kt + 2 < nk) {         // stay 2 tiles ahead of consumption
                const int pm = m0 + (tid & 127);
                if (pm < M)
                    __builtin_prefetch((const void*)&A[aBase + (long long)pm * lda + (kt + 2) * BK], 0, 1);
            }
            stage(1 - cur, (kt + 1) * BK);
        }
        compute(cur);
        __syncthreads();
    }

    // ---- epilogue (C/D layout: VGPR r -> M = r + 8*hi, N = lane%16) -----
#pragma unroll
    for (int mf = 0; mf < 2; ++mf)
#pragma unroll
        for (int nf = 0; nf < 4; ++nf)
#pragma unroll
            for (int r = 0; r < 8; ++r) {
                const int gm = m0 + wm * 32 + mf * 16 + r + 8 * hi;
                const int gn = n0 + wn * 64 + nf * 16 + laneM;
                if (gm < M && gn < N) {
                    float v = acc[mf][nf][r] * alpha;
                    if (HASBIAS) v += bias[gn];
                    if (ACT)     v = (v >= 0.0f) ? v : 0.01f * v;   // LeakyReLU
                    if (HASRES)  v += resid[rBase + (long long)gm * ldr + gn];
                    C[cBase + (long long)gm * ldc + gn] = v;
                }
            }
}

// ---------------------------------------------------------------------------
__global__ __launch_bounds__(256)
void layernorm_rows(const float* __restrict__ in, float* __restrict__ out,
                    const float* __restrict__ gam, const float* __restrict__ bet, int E_)
{
    __shared__ float r1[256], r2[256];
    const int tid = threadIdx.x;
    const long long base = (long long)blockIdx.x * E_;
    float s = 0.0f, q = 0.0f;
    for (int e = tid; e < E_; e += 256) { float v = in[base + e]; s += v; q += v * v; }
    r1[tid] = s; r2[tid] = q; __syncthreads();
    for (int off = 128; off > 0; off >>= 1) {
        if (tid < off) { r1[tid] += r1[tid + off]; r2[tid] += r2[tid + off]; }
        __syncthreads();
    }
    const float mean = r1[0] / (float)E_;
    const float var  = r2[0] / (float)E_ - mean * mean;
    const float rstd = rsqrtf(var + 1e-5f);
    for (int e = tid; e < E_; e += 256)
        out[base + e] = (in[base + e] - mean) * rstd * gam[e] + bet[e];
}

__global__ __launch_bounds__(256)
void softmax_rows(float* __restrict__ sc, int Lr, int causal)
{
    __shared__ float red[256];
    const int tid = threadIdx.x;
    const int row = blockIdx.x;
    const int qq  = row % Lr;
    const long long base = (long long)row * Lr;
    float mx = -3.0e38f;
    for (int c = tid; c < Lr; c += 256) {
        float v = sc[base + c];
        if (causal && c > qq) v = -1.0e12f;
        mx = fmaxf(mx, v);
    }
    red[tid] = mx; __syncthreads();
    for (int off = 128; off > 0; off >>= 1) {
        if (tid < off) red[tid] = fmaxf(red[tid], red[tid + off]);
        __syncthreads();
    }
    mx = red[0]; __syncthreads();
    float sum = 0.0f;
    for (int c = tid; c < Lr; c += 256) {
        float v = sc[base + c];
        if (causal && c > qq) v = -1.0e12f;
        sum += __expf(v - mx);
    }
    red[tid] = sum; __syncthreads();
    for (int off = 128; off > 0; off >>= 1) {
        if (tid < off) red[tid] += red[tid + off];
        __syncthreads();
    }
    const float inv = 1.0f / red[0];
    for (int c = tid; c < Lr; c += 256) {
        float v = sc[base + c];
        if (causal && c > qq) v = -1.0e12f;
        sc[base + c] = __expf(v - mx) * inv;
    }
}

// im2col for Conv2d(C,E,P,stride=P): A[(b*L+l)*768 + (c*256+py*16+px)]
__global__ void im2col_k(const float* __restrict__ x, float* __restrict__ Ap, int total)
{
    int idx = blockIdx.x * blockDim.x + threadIdx.x;
    if (idx >= total) return;
    int k = idx % (CC * PP * PP);
    int r = idx / (CC * PP * PP);
    int l = r % LL, b = r / LL;
    int c = k / (PP * PP);
    int rem = k % (PP * PP);
    int py = rem / PP, px = rem % PP;
    int gy = l / GW, gx = l % GW;
    Ap[idx] = x[(((long long)b * CC + c) * HH + gy * PP + py) * WW + gx * PP + px];
}

// conv_w (E, 768) -> WT (768, E)
__global__ void transpose_w_k(const float* __restrict__ w, float* __restrict__ wt, int total)
{
    int idx = blockIdx.x * blockDim.x + threadIdx.x;
    if (idx >= total) return;
    int k = idx / EE, e = idx % EE;
    wt[idx] = w[(long long)e * (CC * PP * PP) + k];
}

__global__ void add_pos_k(float* __restrict__ h, const float* __restrict__ pos, int total)
{
    int idx = blockIdx.x * blockDim.x + threadIdx.x;
    if (idx >= total) return;
    h[idx] += pos[idx % (LL * EE)];
}

// Y (B,L,CPP) -> out (B,C,H,W) via transpose(0,2,1).reshape
__global__ void rearrange_k(const float* __restrict__ Y, float* __restrict__ out, int total)
{
    int idx = blockIdx.x * blockDim.x + threadIdx.x;
    if (idx >= total) return;
    const int CHW = CC * HH * WW;        // = CPP * L
    int b = idx / CHW, r = idx % CHW;
    int cpp = r / LL, l = r % LL;
    out[idx] = Y[((long long)b * LL + l) * CPPV + cpp];
}

// ---------------------------------------------------------------------------
static inline int cdiv(int a, int b) { return (a + b - 1) / b; }

static void launch_gemm(hipStream_t s,
                        const float* A, int lda, long long aOut, long long aIn,
                        const float* Bp, int ldb, long long bOut, long long bIn, int transB,
                        float* C, int ldc, long long cOut, long long cIn,
                        const float* bias,
                        const float* resid, int ldr, long long rOut, long long rIn,
                        int M, int N, int K, float alpha, int act,
                        int batches, int innerCount)
{
    dim3 grid(cdiv(N, BN), cdiv(M, BM), batches);
    dim3 blk(256);
#define GEMM_ARGS A, lda, aOut, aIn, Bp, ldb, bOut, bIn, C, ldc, cOut, cIn, \
                  bias, resid, ldr, rOut, rIn, M, N, K, alpha, innerCount
    if (transB)
        gemm_wmma_bf16<1, 0, 0, 0><<<grid, blk, 0, s>>>(GEMM_ARGS);   // Q@K^T scores
    else if (act)
        gemm_wmma_bf16<0, 1, 1, 0><<<grid, blk, 0, s>>>(GEMM_ARGS);   // W1 + leaky
    else if (resid)
        gemm_wmma_bf16<0, 0, 1, 1><<<grid, blk, 0, s>>>(GEMM_ARGS);   // Wo / W2 + residual
    else if (bias)
        gemm_wmma_bf16<0, 0, 1, 0><<<grid, blk, 0, s>>>(GEMM_ARGS);   // QKV / patch / proj
    else
        gemm_wmma_bf16<0, 0, 0, 0><<<grid, blk, 0, s>>>(GEMM_ARGS);   // attn A@V
#undef GEMM_ARGS
}

extern "C" void kernel_launch(void* const* d_in, const int* in_sizes, int n_in,
                              void* d_out, int out_size, void* d_ws, size_t ws_size,
                              hipStream_t stream)
{
    const float* x      = (const float*)d_in[0];
    const float* conv_w = (const float*)d_in[1];
    const float* conv_b = (const float*)d_in[2];
    const float* pos    = (const float*)d_in[3];
    const float* ln0_s  = (const float*)d_in[4];
    const float* ln0_b  = (const float*)d_in[5];
    const float* Wq     = (const float*)d_in[6];
    const float* bq     = (const float*)d_in[7];
    const float* Wk     = (const float*)d_in[8];
    const float* bk     = (const float*)d_in[9];
    const float* Wv     = (const float*)d_in[10];
    const float* bv     = (const float*)d_in[11];
    const float* Wo     = (const float*)d_in[12];
    const float* bo     = (const float*)d_in[13];
    const float* ln1_s  = (const float*)d_in[14];
    const float* ln1_b  = (const float*)d_in[15];
    const float* W1     = (const float*)d_in[16];
    const float* b1     = (const float*)d_in[17];
    const float* W2     = (const float*)d_in[18];
    const float* b2     = (const float*)d_in[19];
    const float* proj_w = (const float*)d_in[20];
    const float* proj_b = (const float*)d_in[21];
    float* out = (float*)d_out;

    const int M = BB * LL;                       // 4608 token rows
    const size_t nBLE = (size_t)BB * LL * EE;    // 3,538,944

    float* ws   = (float*)d_ws;
    float* h    = ws;
    float* Xn   = ws + 1 * nBLE;
    float* Qb   = ws + 2 * nBLE;
    float* Kb   = ws + 3 * nBLE;
    float* Vb   = ws + 4 * nBLE;
    float* Octx = ws + 5 * nBLE;                 // (B, NH, L, hd) == faithful (B,L,E) view
    float* Xr   = ws + 6 * nBLE;
    float* Ap   = ws + 7 * nBLE;                 // im2col patches (B*L, 768)
    float* Yb   = ws + 8 * nBLE;                 // final proj output (B,L,CPP)
    float* M1   = ws + 9 * nBLE;                 // MLP hidden (B*L, FF) = 4*nBLE
    float* Sc   = ws + 13 * nBLE;                // scores (B,NH,L,L)
    float* WT   = Sc + (size_t)BB * NHH * LL * LL;  // conv_w transposed (768, E)

    // ---- patch embed as GEMM --------------------------------------------
    {
        int tot = M * CC * PP * PP;
        im2col_k<<<cdiv(tot, 256), 256, 0, stream>>>(x, Ap, tot);
        int tw = CC * PP * PP * EE;
        transpose_w_k<<<cdiv(tw, 256), 256, 0, stream>>>(conv_w, WT, tw);
        launch_gemm(stream, Ap, CC * PP * PP, 0, 0,
                    WT, EE, 0, 0, 0,
                    h, EE, 0, 0,
                    conv_b, nullptr, 0, 0, 0,
                    M, EE, CC * PP * PP, 1.0f, 0, 1, 1);
        int th = (int)nBLE;
        add_pos_k<<<cdiv(th, 256), 256, 0, stream>>>(h, pos, th);
    }

    const long long LE  = (long long)LL * EE;
    const long long LL2 = (long long)LL * LL;

    for (int i = 0; i < NLL; ++i) {
        const float* Wq_i = Wq + (size_t)i * EE * EE;
        const float* Wk_i = Wk + (size_t)i * EE * EE;
        const float* Wv_i = Wv + (size_t)i * EE * EE;
        const float* Wo_i = Wo + (size_t)i * EE * EE;
        const float* W1_i = W1 + (size_t)i * EE * FFF;
        const float* W2_i = W2 + (size_t)i * FFF * EE;

        // ---- attention block ----
        layernorm_rows<<<M, 256, 0, stream>>>(h, Xn, ln0_s + i * EE, ln0_b + i * EE, EE);

        launch_gemm(stream, Xn, EE, 0, 0, Wq_i, EE, 0, 0, 0, Qb, EE, 0, 0,
                    bq + i * EE, nullptr, 0, 0, 0, M, EE, EE, 1.0f, 0, 1, 1);
        launch_gemm(stream, Xn, EE, 0, 0, Wk_i, EE, 0, 0, 0, Kb, EE, 0, 0,
                    bk + i * EE, nullptr, 0, 0, 0, M, EE, EE, 1.0f, 0, 1, 1);
        launch_gemm(stream, Xn, EE, 0, 0, Wv_i, EE, 0, 0, 0, Vb, EE, 0, 0,
                    bv + i * EE, nullptr, 0, 0, 0, M, EE, EE, 1.0f, 0, 1, 1);

        // scores[b,h] = (Q_h @ K_h^T) * hd^-0.5   (96 batched GEMMs)
        launch_gemm(stream, Qb, EE, LE, HD,
                    Kb, EE, LE, HD, /*transB=*/1,
                    Sc, LL, (long long)NHH * LL2, LL2,
                    nullptr, nullptr, 0, 0, 0,
                    LL, LL, HD, 0.125f /* 1/sqrt(64) */, 0, BB * NHH, NHH);

        softmax_rows<<<BB * NHH * LL, 256, 0, stream>>>(Sc, LL, (i == 0) ? 1 : 0);

        // O[b,h] = A @ V_h  -> Octx stored (B,NH,L,hd): faithful reshape to (B,L,E)
        launch_gemm(stream, Sc, LL, (long long)NHH * LL2, LL2,
                    Vb, EE, LE, HD, 0,
                    Octx, HD, LE, (long long)LL * HD,
                    nullptr, nullptr, 0, 0, 0,
                    LL, HD, LL, 1.0f, 0, BB * NHH, NHH);

        // h = Octx @ Wo + bo + h
        launch_gemm(stream, Octx, EE, 0, 0, Wo_i, EE, 0, 0, 0, h, EE, 0, 0,
                    bo + i * EE, h, EE, 0, 0, M, EE, EE, 1.0f, 0, 1, 1);

        // ---- MLP block ----
        layernorm_rows<<<M, 256, 0, stream>>>(h, Xr, ln1_s + i * EE, ln1_b + i * EE, EE);

        launch_gemm(stream, Xr, EE, 0, 0, W1_i, FFF, 0, 0, 0, M1, FFF, 0, 0,
                    b1 + i * FFF, nullptr, 0, 0, 0, M, FFF, EE, 1.0f, /*leaky*/1, 1, 1);

        // h = M1 @ W2 + b2 + Xr  (residual onto post-LN, as in source)
        launch_gemm(stream, M1, FFF, 0, 0, W2_i, EE, 0, 0, 0, h, EE, 0, 0,
                    b2 + i * EE, Xr, EE, 0, 0, M, EE, FFF, 1.0f, 0, 1, 1);
    }

    // ---- output projection + NCHW rearrange -----------------------------
    launch_gemm(stream, h, EE, 0, 0, proj_w, CPPV, 0, 0, 0, Yb, CPPV, 0, 0,
                proj_b, nullptr, 0, 0, 0, M, CPPV, EE, 1.0f, 0, 1, 1);

    int tot = BB * CC * HH * WW;
    rearrange_k<<<cdiv(tot, 256), 256, 0, stream>>>(Yb, out, tot);
}
